// DrowsinessFusionNetwork_79182017069496
// MI455X (gfx1250) — 8-bit matrix operand order repaired and checked by reading, compiled, not run on hardware
//
#include <hip/hip_runtime.h>


typedef __attribute__((ext_vector_type(8))) int          v8i;
typedef __attribute__((ext_vector_type(4))) int          v4i;
typedef __attribute__((ext_vector_type(4))) unsigned int v4u;
typedef unsigned char u8;
typedef unsigned int  u32;

#if defined(__has_builtin)
#if __has_builtin(__builtin_amdgcn_tensor_load_to_lds) && \
    __has_builtin(__builtin_amdgcn_s_wait_tensorcnt)
#define HAVE_TDM 1
#endif
#endif
#ifndef HAVE_TDM
#define HAVE_TDM 0
#endif

__host__ __device__ constexpr int cdiv(int a, int b) { return (a + b - 1) / b; }

__global__ void quant_input_kernel(const float* __restrict__ x,
                                   const float* __restrict__ scale,
                                   signed char* __restrict__ q, int n)
{
    int i = blockIdx.x * blockDim.x + threadIdx.x;
    if (i >= n) return;
    float s = *scale;
    float v = rintf(x[i] / s);
    v = v < -128.f ? -128.f : (v > 127.f ? 127.f : v);
    q[i] = (signed char)v;
}

template<int CIN, int COUT, int KH, int KW>
__global__ void prep_weights_kernel(const float* __restrict__ w,
                                    const float* __restrict__ bias,
                                    const float* __restrict__ gamma,
                                    const float* __restrict__ beta,
                                    const float* __restrict__ mean,
                                    const float* __restrict__ var,
                                    const float* __restrict__ act_scale,
                                    const float* __restrict__ in_scale,
                                    signed char* __restrict__ wpack,
                                    float* __restrict__ coefA,
                                    float* __restrict__ coefB)
{
    constexpr int K  = CIN * KH * KW;
    constexpr int KT = cdiv(K, 64);
    constexpr int MT = cdiv(COUT, 16);
    __shared__ float red[256];
    const int tid = threadIdx.x;

    float m = 0.f;
    for (int i = tid; i < COUT * K; i += 256) m = fmaxf(m, fabsf(w[i]));
    red[tid] = m;
    __syncthreads();
    for (int s = 128; s > 0; s >>= 1) {
        if (tid < s) red[tid] = fmaxf(red[tid], red[tid + s]);
        __syncthreads();
    }
    const float sw = red[0] / 127.f + 1e-12f;

    for (int idx = tid; idx < MT * KT * 32 * 8; idx += 256) {
        int v    = idx & 7;
        int lane = (idx >> 3) & 31;
        int kt   = (idx >> 8) % KT;
        int mt   = (idx >> 8) / KT;
        int c    = mt * 16 + (lane & 15);
        int half = lane >> 4;
        int kbase = kt * 64 + (v >> 1) * 16 + (v & 1) * 4 + half * 8;
        u32 d = 0;
        for (int j = 0; j < 4; ++j) {
            int k = kbase + j;
            int qv = 0;
            if (c < COUT && k < K) {
                float q = rintf(w[c * K + k] / sw);
                q = q < -127.f ? -127.f : (q > 127.f ? 127.f : q);
                qv = (int)q;
            }
            d |= ((u32)(qv & 0xff)) << (8 * j);
        }
        ((u32*)wpack)[idx] = d;
    }

    if (tid < COUT) {
        float inv = gamma[tid] / sqrtf(var[tid] + 1e-5f);
        float sa  = *act_scale;
        coefA[tid] = sw * (*in_scale) * inv / sa;
        coefB[tid] = (bias[tid] * inv + beta[tid] - mean[tid] * inv) / sa;
    }
}

template<int CIN, int COUT, int KH, int KW, int HIN, int WIN,
         int PAD_T, int PAD_B, int PAD_L, int PAD_R, bool BSIGNED>
__global__ __launch_bounds__(256)
void conv_stage_kernel(const u8* __restrict__ actin,
                       const signed char* __restrict__ wpack,
                       const float* __restrict__ coefA,
                       const float* __restrict__ coefB,
                       u8* __restrict__ convout)
{
    constexpr int HOUTC  = HIN + PAD_T + PAD_B - KH + 1;
    constexpr int WOUTC  = WIN + PAD_L + PAD_R - KW + 1;
    constexpr int K      = CIN * KH * KW;
    constexpr int KT     = cdiv(K, 64);
    constexpr int MT     = cdiv(COUT, 16);
    constexpr int NPOS   = HOUTC * WOUTC;
    constexpr int PTILES = cdiv(NPOS, 16);
    constexpr int IN_BYTES = CIN * HIN * WIN;

    __shared__ u8 lin[IN_BYTES];
    const int n = blockIdx.y;

#if HAVE_TDM
    if (threadIdx.x < 32) {
        unsigned long long ga = (unsigned long long)(actin + (size_t)n * IN_BYTES);
        u32 lds_off = (u32)(size_t)&lin[0];
        v4u g0;
        g0[0] = 1u;
        g0[1] = lds_off;
        g0[2] = (u32)ga;
        g0[3] = (u32)((ga >> 32) & 0x1FFFFFFull) | (2u << 30);
        v8i g1 = {0, 0, 0, 0, 0, 0, 0, 0};
        g1[0] = 0;
        g1[1] = (int)(((u32)IN_BYTES & 0xFFFFu) << 16);
        g1[2] = (int)(((u32)IN_BYTES >> 16) & 0xFFFFu);
        g1[3] = (int)(((u32)IN_BYTES & 0xFFFFu) << 16);
        g1[5] = (int)IN_BYTES;
        v4i g2 = {0, 0, 0, 0};
        v4i g3 = {0, 0, 0, 0};
        v8i g4 = {0, 0, 0, 0, 0, 0, 0, 0};
        __builtin_amdgcn_tensor_load_to_lds(g0, g1, g2, g3, g4, 0);
        __builtin_amdgcn_s_wait_tensorcnt(0);
    }
#else
    for (int i = threadIdx.x; i < IN_BYTES; i += 256)
        lin[i] = actin[(size_t)n * IN_BYTES + i];
#endif
    if (threadIdx.x == 0) __builtin_prefetch(wpack, 0, 0);
    __syncthreads();

    const int wave = threadIdx.x >> 5;
    const int lane = threadIdx.x & 31;
    const int t    = blockIdx.x * 8 + wave;
    if (t >= MT * PTILES) return;
    const int mt = t / PTILES;
    const int pt = t % PTILES;

    const int N      = lane & 15;
    const int khalf  = lane >> 4;
    const int pos    = pt * 16 + N;
    const bool pvalid = pos < NPOS;
    const int oh = pvalid ? (pos / WOUTC) : 0;
    const int ow = pvalid ? (pos % WOUTC) : 0;

    v8i acc = {0, 0, 0, 0, 0, 0, 0, 0};
    for (int kt = 0; kt < KT; ++kt) {
        v8i a = ((const v8i*)wpack)[(mt * KT + kt) * 32 + lane];

        v8i b;
#pragma unroll
        for (int v = 0; v < 8; ++v) {
            u32 d = 0;
#pragma unroll
            for (int j = 0; j < 4; ++j) {
                int k = kt * 64 + (v >> 1) * 16 + (v & 1) * 4 + khalf * 8 + j;
                u32 byte = 0;
                if (pvalid && k < K) {
                    int cin = k / (KH * KW);
                    int r   = k % (KH * KW);
                    int kh  = r / KW;
                    int kw  = r % KW;
                    int ih  = oh + kh - PAD_T;
                    int iw  = ow + kw - PAD_L;
                    if (ih >= 0 && ih < HIN && iw >= 0 && iw < WIN)
                        byte = lin[(cin * HIN + ih) * WIN + iw];
                }
                d |= byte << (8 * j);
            }
            b[v] = (int)d;
        }
        acc = __builtin_amdgcn_wmma_i32_16x16x64_iu8(true, a, BSIGNED, b, acc,
                                                     false, false);
    }

#pragma unroll
    for (int r = 0; r < 8; ++r) {
        int c = mt * 16 + r + khalf * 8;
        if (c < COUT && pvalid) {
            float y = (float)acc[r] * coefA[c] + coefB[c];
            float q = rintf(y);
            q = q < 0.f ? 0.f : (q > 255.f ? 255.f : q);
            convout[((size_t)n * COUT + c) * NPOS + pos] = (u8)q;
        }
    }
}

template<int C, int HI, int WI, int PH, int PW>
__global__ void maxpool_kernel(const u8* __restrict__ in, u8* __restrict__ out, int B)
{
    constexpr int HO = (HI - PH) / PH + 1;
    constexpr int WO = (WI - PW) / PW + 1;
    int idx = blockIdx.x * blockDim.x + threadIdx.x;
    int total = B * C * HO * WO;
    if (idx >= total) return;
    int wo = idx % WO;
    int ho = (idx / WO) % HO;
    int c  = (idx / (WO * HO)) % C;
    int n  = idx / (WO * HO * C);
    const u8* p = in + ((size_t)n * C + c) * HI * WI;
    u8 m = 0;
    for (int i = 0; i < PH; ++i)
        for (int j = 0; j < PW; ++j) {
            u8 v = p[(ho * PH + i) * WI + (wo * PW + j)];
            m = v > m ? v : m;
        }
    out[idx] = m;
}

__global__ void head_kernel(const u8* __restrict__ eeg_act,
                            const u8* __restrict__ ppg_act,
                            const float* __restrict__ eeg_as,
                            const float* __restrict__ ppg_as,
                            const float* __restrict__ cw,
                            const float* __restrict__ cb,
                            const float* __restrict__ cscale,
                            float* __restrict__ out, int B)
{
    int n = blockIdx.x * blockDim.x + threadIdx.x;
    if (n >= B) return;
    float f[48];
    const float se = *eeg_as, sp = *ppg_as, sc = *cscale;
    for (int c = 0; c < 16; ++c) {
        const u8* p = eeg_act + ((size_t)n * 16 + c) * 34;
        int s = 0;
        for (int i = 0; i < 34; ++i) s += p[i];
        f[c] = ((float)s / 34.f) * se;
    }
    for (int c = 0; c < 32; ++c) {
        const u8* p = ppg_act + ((size_t)n * 32 + c) * 68;
        int s = 0;
        for (int i = 0; i < 68; ++i) s += p[i];
        f[16 + c] = ((float)s / 68.f) * sp;
    }
    for (int i = 0; i < 48; ++i) {
        float q = rintf(f[i] / sc);
        q = q < -128.f ? -128.f : (q > 127.f ? 127.f : q);
        f[i] = q * sc;
    }
    float m = 0.f;
    for (int i = 0; i < 96; ++i) m = fmaxf(m, fabsf(cw[i]));
    const float sw = m / 127.f + 1e-12f;
    for (int o = 0; o < 2; ++o) {
        float acc = cb[o];
        for (int i = 0; i < 48; ++i) {
            float q = rintf(cw[o * 48 + i] / sw);
            q = q < -127.f ? -127.f : (q > 127.f ? 127.f : q);
            acc += f[i] * q * sw;
        }
        out[n * 2 + o] = acc;
    }
}

extern "C" void kernel_launch(void* const* d_in, const int* in_sizes, int n_in,
                              void* d_out, int out_size, void* d_ws, size_t ws_size,
                              hipStream_t stream)
{
    (void)n_in; (void)out_size; (void)ws_size;
    const float* eeg_x = (const float*)d_in[0];
    const float* ppg_x = (const float*)d_in[1];
    const int B = in_sizes[0] / (8 * 2200);

    char* ws = (char*)d_ws;
    auto slotp = [&](int s) -> signed char* { return (signed char*)(ws + (size_t)s * 36864); };
    auto cA    = [&](int s) -> float*       { return (float*)(ws + (size_t)s * 36864 + 32768); };
    auto cB    = [&](int s) -> float*       { return (float*)(ws + (size_t)s * 36864 + 34816); };

    const size_t OFF_ACT_A = 10u * 36864u;
    const size_t OFF_ACT_B = OFF_ACT_A + 10485760u;
    const size_t OFF_TMP   = OFF_ACT_B + 10485760u;
    const size_t OFF_PPGQ  = OFF_TMP   + 37748736u;
    const size_t OFF_EEGF  = OFF_PPGQ  + 2359296u;
    const size_t OFF_PPGF  = OFF_EEGF  + 294912u;

    u8* actA = (u8*)(ws + OFF_ACT_A);
    u8* actB = (u8*)(ws + OFF_ACT_B);
    u8* tmp  = (u8*)(ws + OFF_TMP);
    u8* ppgq = (u8*)(ws + OFF_PPGQ);
    u8* eegf = (u8*)(ws + OFF_EEGF);
    u8* ppgf = (u8*)(ws + OFF_PPGF);

    {
        int ne = B * 8 * 2200;
        quant_input_kernel<<<cdiv(ne, 256), 256, 0, stream>>>(
            eeg_x, (const float*)d_in[2], (signed char*)actA, ne);
        int np = B * 2 * 2200;
        quant_input_kernel<<<cdiv(np, 256), 256, 0, stream>>>(
            ppg_x, (const float*)d_in[38], (signed char*)ppgq, np);
    }

#define PREP(CIN, COUT, KH, KW, base, insc, slot)                                  \
    prep_weights_kernel<CIN, COUT, KH, KW><<<1, 256, 0, stream>>>(                 \
        (const float*)d_in[(base)],     (const float*)d_in[(base) + 1],            \
        (const float*)d_in[(base) + 2], (const float*)d_in[(base) + 3],            \
        (const float*)d_in[(base) + 4], (const float*)d_in[(base) + 5],            \
        (const float*)d_in[(base) + 6], (const float*)d_in[(insc)],                \
        slotp(slot), cA(slot), cB(slot))

#define CONV(CIN, COUT, KH, KW, HIN, WIN, PT_, PB_, PL_, PR_, BS, pin, pout, slot) \
    do {                                                                           \
        constexpr int HOUTC_ = (HIN) + (PT_) + (PB_) - (KH) + 1;                   \
        constexpr int WOUTC_ = (WIN) + (PL_) + (PR_) - (KW) + 1;                   \
        constexpr int MT_    = ((COUT) + 15) / 16;                                 \
        int tiles_ = MT_ * cdiv(HOUTC_ * WOUTC_, 16);                              \
        dim3 g_(cdiv(tiles_, 8), B, 1);                                            \
        conv_stage_kernel<CIN, COUT, KH, KW, HIN, WIN, PT_, PB_, PL_, PR_, BS>     \
            <<<g_, 256, 0, stream>>>((const u8*)(pin), slotp(slot), cA(slot),      \
                                     cB(slot), (u8*)(pout));                       \
    } while (0)

#define POOL(C, HI, WI, PH, PW, pin, pout)                                         \
    do {                                                                           \
        constexpr int HO_ = ((HI) - (PH)) / (PH) + 1;                              \
        constexpr int WO_ = ((WI) - (PW)) / (PW) + 1;                              \
        int total_ = B * (C) * HO_ * WO_;                                          \
        maxpool_kernel<C, HI, WI, PH, PW><<<cdiv(total_, 256), 256, 0, stream>>>(  \
            (const u8*)(pin), (u8*)(pout), B);                                     \
    } while (0)

    PREP(1, 4, 1, 4,     3,  2, 0);
    PREP(4, 16, 1, 16,  10,  9, 1);
    PREP(16, 16, 1, 8,  17, 16, 2);
    PREP(16, 16, 16, 1, 24, 23, 3);
    PREP(16, 16, 8, 1,  31, 30, 4);
    PREP(1, 4, 1, 4,    39, 38, 5);
    PREP(4, 8, 1, 4,    46, 45, 6);
    PREP(8, 16, 1, 4,   53, 52, 7);
    PREP(16, 32, 1, 8,  60, 59, 8);
    PREP(32, 32, 1, 16, 67, 66, 9);

    CONV(1, 4, 1, 4, 8, 2200, 0, 0, 1, 2, true, actA, tmp, 0);
    POOL(4, 8, 2200, 1, 8, tmp, actB);
    CONV(4, 16, 1, 16, 8, 275, 0, 0, 7, 8, false, actB, tmp, 1);
    POOL(16, 8, 275, 1, 4, tmp, actA);
    CONV(16, 16, 1, 8, 8, 68, 0, 0, 3, 4, false, actA, tmp, 2);
    POOL(16, 8, 68, 1, 4, tmp, actB);
    CONV(16, 16, 16, 1, 8, 17, 7, 8, 0, 0, false, actB, tmp, 3);
    POOL(16, 8, 17, 4, 1, tmp, actA);
    CONV(16, 16, 8, 1, 2, 17, 3, 4, 0, 0, false, actA, eegf, 4);

    CONV(1, 4, 1, 4, 2, 2200, 0, 0, 1, 2, true, ppgq, tmp, 5);
    POOL(4, 2, 2200, 1, 4, tmp, actA);
    CONV(4, 8, 1, 4, 2, 550, 0, 0, 1, 2, false, actA, tmp, 6);
    POOL(8, 2, 550, 1, 4, tmp, actB);
    CONV(8, 16, 1, 4, 2, 137, 0, 0, 1, 2, false, actB, tmp, 7);
    POOL(16, 2, 137, 1, 4, tmp, actA);
    CONV(16, 32, 1, 8, 2, 34, 0, 0, 3, 4, false, actA, actB, 8);
    CONV(32, 32, 1, 16, 2, 34, 0, 0, 7, 8, false, actB, ppgf, 9);

    head_kernel<<<cdiv(B, 128), 128, 0, stream>>>(
        eegf, ppgf,
        (const float*)d_in[37], (const float*)d_in[73],
        (const float*)d_in[74], (const float*)d_in[75], (const float*)d_in[76],
        (float*)d_out, B);

#undef PREP
#undef CONV
#undef POOL
}
